// sLSTMLayer_46840913330315
// MI455X (gfx1250) — matrix operand order repaired and checked by reading, not compiled, not run on hardware
//
#include <hip/hip_runtime.h>
#include <hip/hip_bf16.h>

// ---------------------------------------------------------------------------
// sLSTM (2-layer, exp input gate, LayerNorm) for MI455X / gfx1250.
// bf16 WMMA (v_wmma_f32_16x16x32_bf16) for all GEMMs, fp32 accumulate,
// fp32 elementwise + LayerNorms. Weights pre-swizzled into WMMA B-fragment
// layout so B loads are contiguous b128s; weights stay L2-resident across
// the 512 sequential steps. K-loop unrolled x2 with independent fragment
// registers to hide WMMA->VALU WAR hazards; pointer-increment addressing.
// ---------------------------------------------------------------------------

typedef __attribute__((ext_vector_type(16))) __bf16 v16bf;
typedef __attribute__((ext_vector_type(8)))  float  v8f;

#define T_STEPS 512
#define BATCH   64
#define IN_DIM  256
#define HID     1024
#define G4      (4 * HID)     // 4096
#define LN_EPSF 1e-5f

union AFrag { v16bf v; uint4 u[2]; };

__device__ __forceinline__ float sigf(float x) { return 1.0f / (1.0f + __expf(-x)); }

__device__ __forceinline__ void load_a(AFrag& f, const __bf16* p)
{
    const uint4* q = (const uint4*)p;
    f.u[0] = q[0];      // K offsets +0..7   (this lane-half)
    f.u[1] = q[2];      // K offsets +16..23 (this lane-half)
}

__device__ __forceinline__ void load_b(AFrag& f, const __bf16* p)
{
    const uint4* q = (const uint4*)p;
    f.u[0] = q[0];
    f.u[1] = q[1];
}

// ---------------------------------------------------------------------------
// 64x64 output tile GEMM: out[0:64, ct*64 : ct*64+64] (+= two K-phases)
// Block = 256 threads = 8 waves. Wave w: row-tile w>>1, col-tiles {2*(w&1), +1}.
// A: row-major bf16 [64 x K].  W*: pre-swizzled bf16 fragments.
// ---------------------------------------------------------------------------
__device__ void gemm_tile_64x64(const __bf16* __restrict__ A1, int K1,
                                const __bf16* __restrict__ W1sw,
                                const __bf16* __restrict__ A2, int K2,
                                const __bf16* __restrict__ W2sw,
                                const float*  __restrict__ bias,
                                float* __restrict__ out, int N, int colTile64)
{
    const int tid   = threadIdx.x;
    const int wave  = tid >> 5;
    const int lane  = tid & 31;
    const int rt    = wave >> 1;                       // 0..3 row tile
    const int ct0   = colTile64 * 4 + (wave & 1) * 2;  // global 16-wide col tile
    const int n16   = N >> 4;
    const int lmod  = lane & 15;
    const int lhalf = lane >> 4;
    const int m     = (rt << 4) + lmod;                // A row for this lane
    const size_t bstride = (size_t)n16 * 512;          // bf16 elems per k-tile of W

    v8f acc0 = {0, 0, 0, 0, 0, 0, 0, 0};
    v8f acc1 = {0, 0, 0, 0, 0, 0, 0, 0};

    for (int phase = 0; phase < 2; ++phase) {
        const __bf16* A = phase ? A2 : A1;
        const __bf16* W = phase ? W2sw : W1sw;
        const int     K = phase ? K2 : K1;
        if (A == nullptr || K == 0) continue;

        const __bf16* ap = A + (size_t)m * K + lhalf * 8;        // +32 per k-tile
        const __bf16* bp = W + ((size_t)ct0 * 32 + lane) * 16;   // +bstride per k-tile
        const int nkt = K >> 5;

        // unroll x2: independent fragment registers hide WMMA->VALU hazards
        for (int kt = 0; kt < nkt; kt += 2) {
            AFrag a0, a1, b00, b01, b10, b11;
            load_a(a0, ap);
            load_a(a1, ap + 32);
            load_b(b00, bp);
            load_b(b01, bp + 512);
            load_b(b10, bp + bstride);
            load_b(b11, bp + bstride + 512);
            acc0 = __builtin_amdgcn_wmma_f32_16x16x32_bf16(false, a0.v, false, b00.v,
                                                           (short)0, acc0, false, false);
            acc1 = __builtin_amdgcn_wmma_f32_16x16x32_bf16(false, a0.v, false, b01.v,
                                                           (short)0, acc1, false, false);
            acc0 = __builtin_amdgcn_wmma_f32_16x16x32_bf16(false, a1.v, false, b10.v,
                                                           (short)0, acc0, false, false);
            acc1 = __builtin_amdgcn_wmma_f32_16x16x32_bf16(false, a1.v, false, b11.v,
                                                           (short)0, acc1, false, false);
            ap += 64;
            bp += 2 * bstride;
        }
    }

    // C/D layout: VGPR r, lanes 0-15 -> M=r, lanes 16-31 -> M=8+r; N = lane&15
    const int row0 = (rt << 4) + (lhalf << 3);
    const int col0 = (ct0 << 4) + lmod;
    float bias0 = 0.f, bias1 = 0.f;
    if (bias) { bias0 = bias[col0]; bias1 = bias[col0 + 16]; }
    float* op = out + (size_t)row0 * N + col0;
#pragma unroll
    for (int r = 0; r < 8; ++r) {
        op[0]  = acc0[r] + bias0;
        op[16] = acc1[r] + bias1;
        op += N;
    }
}

// gates0 = xt@Wx0 + h0@Wh0 + b0 (blocks 0..63), R = xt@Wproj (blocks 64..79)
__global__ void __launch_bounds__(256)
step_gemm0_kernel(const __bf16* __restrict__ Xt, const __bf16* __restrict__ Wx0sw,
                  const __bf16* __restrict__ h0b, const __bf16* __restrict__ Wh0sw,
                  const __bf16* __restrict__ Wprojsw, const float* __restrict__ b0,
                  float* __restrict__ gates0, float* __restrict__ R)
{
    const int blk = blockIdx.x;
    if (blk < 64)
        gemm_tile_64x64(Xt, IN_DIM, Wx0sw, h0b, HID, Wh0sw, b0, gates0, G4, blk);
    else
        gemm_tile_64x64(Xt, IN_DIM, Wprojsw, nullptr, 0, nullptr, nullptr, R, HID, blk - 64);
}

// gates1 = h0@Wx1 + h1@Wh1 + b1
__global__ void __launch_bounds__(256)
step_gemm1_kernel(const __bf16* __restrict__ h0b, const __bf16* __restrict__ Wx1sw,
                  const __bf16* __restrict__ h1b, const __bf16* __restrict__ Wh1sw,
                  const float* __restrict__ b1, float* __restrict__ gates1)
{
    gemm_tile_64x64(h0b, HID, Wx1sw, h1b, HID, Wh1sw, b1, gates1, G4, blockIdx.x);
}

// ---------------------------------------------------------------------------
// Block-wide (256 thr) two-value reduction: wave32 shuffles + LDS combine.
// ---------------------------------------------------------------------------
__device__ __forceinline__ float2 blk_reduce2(float s, float q)
{
    __shared__ float red[16];
#pragma unroll
    for (int off = 16; off > 0; off >>= 1) {
        s += __shfl_xor(s, off, 32);
        q += __shfl_xor(q, off, 32);
    }
    const int wave = threadIdx.x >> 5;
    const int lane = threadIdx.x & 31;
    __syncthreads();
    if (lane == 0) { red[wave] = s; red[wave + 8] = q; }
    __syncthreads();
    s = 0.f; q = 0.f;
#pragma unroll
    for (int i = 0; i < 8; ++i) { s += red[i]; q += red[i + 8]; }
    return make_float2(s, q);
}

// ---------------------------------------------------------------------------
// sLSTM cell + 3 LayerNorms for one batch row per block (H=1024, 4 elem/thr).
//   c = LN( sigmoid(f)*c + exp(i)*tanh(g) )
//   h = LN( sigmoid(o)*tanh(c) )
//   h_out = LN( h + resid )
// ---------------------------------------------------------------------------
__global__ void __launch_bounds__(256)
cell_kernel(const float* __restrict__ gates, const float* __restrict__ resid,
            float* __restrict__ c_state, float* __restrict__ h_state,
            __bf16* __restrict__ h_b16,
            const float* __restrict__ lnc_g, const float* __restrict__ lnc_b,
            const float* __restrict__ lnh_g, const float* __restrict__ lnh_b,
            const float* __restrict__ lno_g, const float* __restrict__ lno_b,
            float* __restrict__ out)
{
    const int b = blockIdx.x;
    const int j = threadIdx.x * 4;
    const float* g = gates + (size_t)b * G4;
    const size_t rowBase = (size_t)b * HID + j;

    float cr[4], ov[4];
    float s = 0.f, q = 0.f;
#pragma unroll
    for (int k = 0; k < 4; ++k) {
        const float iv = g[j + k];
        const float fv = g[HID + j + k];
        const float gv = g[2 * HID + j + k];
        ov[k] = g[3 * HID + j + k];
        const float cold = c_state[rowBase + k];
        const float v = sigf(fv) * cold + __expf(iv) * tanhf(gv);
        cr[k] = v; s += v; q += v * v;
    }
    float2 r = blk_reduce2(s, q);
    float mean = r.x * (1.f / HID);
    float inv  = rsqrtf(fmaxf(r.y * (1.f / HID) - mean * mean, 0.f) + LN_EPSF);

    float hr[4];
    s = 0.f; q = 0.f;
#pragma unroll
    for (int k = 0; k < 4; ++k) {
        const float cn = (cr[k] - mean) * inv * lnc_g[j + k] + lnc_b[j + k];
        c_state[rowBase + k] = cn;
        const float hv = sigf(ov[k]) * tanhf(cn);
        hr[k] = hv; s += hv; q += hv * hv;
    }
    r = blk_reduce2(s, q);
    mean = r.x * (1.f / HID);
    inv  = rsqrtf(fmaxf(r.y * (1.f / HID) - mean * mean, 0.f) + LN_EPSF);

    float rv[4];
    s = 0.f; q = 0.f;
#pragma unroll
    for (int k = 0; k < 4; ++k) {
        const float hln = (hr[k] - mean) * inv * lnh_g[j + k] + lnh_b[j + k];
        const float v = hln + resid[rowBase + k];
        rv[k] = v; s += v; q += v * v;
    }
    r = blk_reduce2(s, q);
    mean = r.x * (1.f / HID);
    inv  = rsqrtf(fmaxf(r.y * (1.f / HID) - mean * mean, 0.f) + LN_EPSF);

#pragma unroll
    for (int k = 0; k < 4; ++k) {
        const float ho = (rv[k] - mean) * inv * lno_g[j + k] + lno_b[j + k];
        h_state[rowBase + k] = ho;
        h_b16[rowBase + k]   = (__bf16)ho;
        if (out) out[rowBase + k] = ho;
    }
}

// ---------------------------------------------------------------------------
// Prep kernels
// ---------------------------------------------------------------------------
// Swizzle fp32 row-major [K x N] weights into bf16 WMMA-B fragment layout:
// dst[((kt*(N/16)+nt)*32 + lane)*16 + e] = W[(kt*32 + koff) * N + nt*16 + (lane&15)]
// koff = (e>>3)*16 + (lane>>4)*8 + (e&7)   (32x16 bf16 B: lanes 0-15 hold K 0..7,16..23; 16-31 hold 8..15,24..31)
__global__ void swizzleW_kernel(const float* __restrict__ src, __bf16* __restrict__ dst,
                                int K, int N)
{
    const int i = blockIdx.x * blockDim.x + threadIdx.x;
    if (i >= K * N) return;
    const int e   = i & 15;
    const int L   = (i >> 4) & 31;
    const int blk = i >> 9;
    const int n16 = N >> 4;
    const int nt  = blk % n16;
    const int kt  = blk / n16;
    const int koff = ((e >> 3) << 4) + ((L >> 4) << 3) + (e & 7);
    const int srcK = (kt << 5) + koff;
    const int srcN = (nt << 4) + (L & 15);
    dst[i] = (__bf16)src[(size_t)srcK * N + srcN];
}

__global__ void cvt_bf16_kernel(const float* __restrict__ src, __bf16* __restrict__ dst, int n)
{
    const int i = blockIdx.x * blockDim.x + threadIdx.x;
    if (i < n) dst[i] = (__bf16)src[i];
}

__global__ void zero_state_kernel(float* h0, float* c0, float* h1, float* c1,
                                  __bf16* h0b, __bf16* h1b)
{
    const int i = blockIdx.x * blockDim.x + threadIdx.x;
    if (i < BATCH * HID) {
        h0[i] = 0.f; c0[i] = 0.f; h1[i] = 0.f; c1[i] = 0.f;
        h0b[i] = (__bf16)0.f; h1b[i] = (__bf16)0.f;
    }
}

// d_out tail: [h0 | h1 | c0 | c1], each BATCH*HID floats
__global__ void finalize_kernel(const float* __restrict__ h0, const float* __restrict__ h1,
                                const float* __restrict__ c0, const float* __restrict__ c1,
                                float* __restrict__ tail)
{
    const int i = blockIdx.x * blockDim.x + threadIdx.x;
    const int n = BATCH * HID;
    if (i >= 4 * n) return;
    float v;
    if (i < n)          v = h0[i];
    else if (i < 2 * n) v = h1[i - n];
    else if (i < 3 * n) v = c0[i - 2 * n];
    else                v = c1[i - 3 * n];
    tail[i] = v;
}

// ---------------------------------------------------------------------------
// Launch
// ---------------------------------------------------------------------------
extern "C" void kernel_launch(void* const* d_in, const int* in_sizes, int n_in,
                              void* d_out, int out_size, void* d_ws, size_t ws_size,
                              hipStream_t stream)
{
    const float* x     = (const float*)d_in[0];
    const float* Wproj = (const float*)d_in[1];
    const float* Wx0   = (const float*)d_in[2];
    const float* Wh0   = (const float*)d_in[3];
    const float* b0    = (const float*)d_in[4];
    const float* Wx1   = (const float*)d_in[5];
    const float* Wh1   = (const float*)d_in[6];
    const float* b1    = (const float*)d_in[7];
    const float* lnc_g = (const float*)d_in[8];
    const float* lnc_b = (const float*)d_in[9];
    const float* lnh_g = (const float*)d_in[10];
    const float* lnh_b = (const float*)d_in[11];
    const float* lno_g = (const float*)d_in[12];
    const float* lno_b = (const float*)d_in[13];
    float* out = (float*)d_out;

    // workspace carve-out (256B aligned)
    char* p = (char*)d_ws;
    auto take = [&](size_t bytes) { char* r = p; p += (bytes + 255) & ~(size_t)255; return r; };
    __bf16* Wx0sw   = (__bf16*)take((size_t)IN_DIM * G4 * 2);
    __bf16* Wh0sw   = (__bf16*)take((size_t)HID * G4 * 2);
    __bf16* Wprojsw = (__bf16*)take((size_t)IN_DIM * HID * 2);
    __bf16* Wx1sw   = (__bf16*)take((size_t)HID * G4 * 2);
    __bf16* Wh1sw   = (__bf16*)take((size_t)HID * G4 * 2);
    __bf16* Xb      = (__bf16*)take((size_t)T_STEPS * BATCH * IN_DIM * 2);
    float*  gates0  = (float*)take((size_t)BATCH * G4 * 4);
    float*  gates1  = (float*)take((size_t)BATCH * G4 * 4);
    float*  R       = (float*)take((size_t)BATCH * HID * 4);
    float*  h0      = (float*)take((size_t)BATCH * HID * 4);
    float*  c0      = (float*)take((size_t)BATCH * HID * 4);
    float*  h1      = (float*)take((size_t)BATCH * HID * 4);
    float*  c1      = (float*)take((size_t)BATCH * HID * 4);
    __bf16* h0b     = (__bf16*)take((size_t)BATCH * HID * 2);
    __bf16* h1b     = (__bf16*)take((size_t)BATCH * HID * 2);

    // one-time (per launch) prep: weight swizzle + x conversion + state init
    int n;
    n = IN_DIM * G4;  swizzleW_kernel<<<(n + 255) / 256, 256, 0, stream>>>(Wx0, Wx0sw, IN_DIM, G4);
    n = HID * G4;     swizzleW_kernel<<<(n + 255) / 256, 256, 0, stream>>>(Wh0, Wh0sw, HID, G4);
    n = IN_DIM * HID; swizzleW_kernel<<<(n + 255) / 256, 256, 0, stream>>>(Wproj, Wprojsw, IN_DIM, HID);
    n = HID * G4;     swizzleW_kernel<<<(n + 255) / 256, 256, 0, stream>>>(Wx1, Wx1sw, HID, G4);
    n = HID * G4;     swizzleW_kernel<<<(n + 255) / 256, 256, 0, stream>>>(Wh1, Wh1sw, HID, G4);
    n = T_STEPS * BATCH * IN_DIM;
    cvt_bf16_kernel<<<(n + 255) / 256, 256, 0, stream>>>(x, Xb, n);
    n = BATCH * HID;
    zero_state_kernel<<<(n + 255) / 256, 256, 0, stream>>>(h0, c0, h1, c1, h0b, h1b);

    // sequential recurrence
    for (int t = 0; t < T_STEPS; ++t) {
        const __bf16* Xt = Xb + (size_t)t * BATCH * IN_DIM;
        step_gemm0_kernel<<<80, 256, 0, stream>>>(Xt, Wx0sw, h0b, Wh0sw, Wprojsw, b0, gates0, R);
        cell_kernel<<<64, 256, 0, stream>>>(gates0, R, c0, h0, h0b,
                                            lnc_g, lnc_b, lnh_g, lnh_b, lno_g, lno_b, nullptr);
        step_gemm1_kernel<<<64, 256, 0, stream>>>(h0b, Wx1sw, h1b, Wh1sw, b1, gates1);
        cell_kernel<<<64, 256, 0, stream>>>(gates1, h0, c1, h1, h1b,
                                            lnc_g + HID, lnc_b + HID, lnh_g + HID, lnh_b + HID,
                                            lno_g + HID, lno_b + HID,
                                            out + (size_t)t * BATCH * HID);
    }

    n = 4 * BATCH * HID;
    finalize_kernel<<<(n + 255) / 256, 256, 0, stream>>>(h0, h1, c0, c1,
                                                         out + (size_t)T_STEPS * BATCH * HID);
}
